// HardParameterSharingModel_58987080843692
// MI455X (gfx1250) — matrix operand order repaired and checked by reading, not compiled, not run on hardware
//
#include <hip/hip_runtime.h>
#include <math.h>

// Model dims
#define HD    512          // hidden
#define G4    2048         // 4*H (gates)
#define BB    32           // batch
#define ETT   256          // encoder T
#define DTT   16           // decoder T
#define HFIN  64
#define LFIN  32
#define OUTD  8
#define GS    260          // LDS gate-row stride (padded: 32x256 tile, no bank conflicts)

typedef __attribute__((ext_vector_type(16))) _Float16 v16h;
typedef __attribute__((ext_vector_type(8)))  _Float16 h8;
typedef __attribute__((ext_vector_type(8)))  float    v8f;

// ---------------------------------------------------------------------------
// WMMA fragment loaders (CDNA5 wave32 layouts, CDNA5 ISA §7.12.2)
// A: 16x32 f16 (MxK).  lane m = lane%16, half = lane/16.
//    VGPR0-3 : K = half*8 + 0..7   -> one contiguous 16B load
//    VGPR4-7 : K = 16 + half*8 + 0..7
// B: 32x16 f16 (KxN), col n = lane%16, K = half*8 + 0..7, then 16 + half*8 + 0..7 (same as A).
//    Weight stored row-major (N,K) => column n of B is row n of W.
// C: 16x16 f32, VGPR r, lanes0-15: M=r,N=lane ; lanes16-31: M=8+r,N=lane-16.
// ---------------------------------------------------------------------------
__device__ inline v16h load_a_frag(const _Float16* A, int lda, int m0, int k0, int lane) {
  int m  = m0 + (lane & 15);
  int hh = lane >> 4;
  const _Float16* p = A + (size_t)m * lda + k0 + hh * 8;
  h8 lo = *(const h8*)(p);
  h8 hi = *(const h8*)(p + 16);
  v16h a;
#pragma unroll
  for (int i = 0; i < 8; ++i) { a[i] = lo[i]; a[i + 8] = hi[i]; }
  return a;
}

__device__ inline v16h load_b_frag(const _Float16* W, int ldw, int n0, int k0, int lane) {
  int n  = n0 + (lane & 15);
  int kh = lane >> 4;
  const _Float16* p = W + (size_t)n * ldw + k0 + kh * 8;
  h8 lo = *(const h8*)(p);
  h8 hi = *(const h8*)(p + 16);
  v16h b;
#pragma unroll
  for (int i = 0; i < 8; ++i) { b[i] = lo[i]; b[i + 8] = hi[i]; }
  return b;
}

__device__ inline v8f wmma32(v16h a, v16h b, v8f c) {
  return __builtin_amdgcn_wmma_f32_16x16x32_f16(false, a, false, b, (short)0, c, false, false);
}

// ---------------------------------------------------------------------------
// Double-buffered 16x64 tile accumulation over K.  The sched_barrier keeps
// the next-k load group strictly before the current-k WMMA group, so the
// WMMAs depend only on loads issued one iteration earlier: backend can use
// a partial s_wait_loadcnt (10 loads in flight) instead of stalling to 0
// before every WMMA.
// ---------------------------------------------------------------------------
__device__ inline void gemm_tile_acc(const _Float16* __restrict__ A, int lda,
                                     const _Float16* __restrict__ W, int ldw,
                                     int m0, int n0, int K, int lane, v8f acc[4])
{
  v16h af = load_a_frag(A, lda, m0, 0, lane);
  v16h bf[4];
#pragma unroll
  for (int s = 0; s < 4; ++s) bf[s] = load_b_frag(W, ldw, n0 + (s << 4), 0, lane);

  for (int k0 = 32; k0 < K; k0 += 32) {
    v16h afn = load_a_frag(A, lda, m0, k0, lane);
    v16h bfn[4];
#pragma unroll
    for (int s = 0; s < 4; ++s) bfn[s] = load_b_frag(W, ldw, n0 + (s << 4), k0, lane);
    __builtin_amdgcn_sched_barrier(0);   // loads above, WMMAs below
#pragma unroll
    for (int s = 0; s < 4; ++s) acc[s] = wmma32(af, bf[s], acc[s]);
    af = afn;
#pragma unroll
    for (int s = 0; s < 4; ++s) bf[s] = bfn[s];
  }
  __builtin_amdgcn_sched_barrier(0);
#pragma unroll
  for (int s = 0; s < 4; ++s) acc[s] = wmma32(af, bf[s], acc[s]);
}

__device__ inline float sigmf(float x) { return 1.f / (1.f + expf(-x)); }

// ---------------------------------------------------------------------------
// Grid-wide barrier (persistent-kernel sync, 8 resident blocks)
// ---------------------------------------------------------------------------
__device__ inline void gsync(unsigned* bar, unsigned nb, int* phase) {
  __threadfence();
  __syncthreads();
  ++(*phase);
  if (threadIdx.x == 0) {
    atomicAdd(bar, 1u);
    unsigned target = (unsigned)(*phase) * nb;
    while (__atomic_load_n(bar, __ATOMIC_RELAXED) < target) {
      __builtin_amdgcn_s_sleep(1);
    }
  }
  __syncthreads();
  __threadfence();
}

// ---------------------------------------------------------------------------
// f32 -> f16 conversion
// ---------------------------------------------------------------------------
__global__ void cvt_f32_f16(const float* __restrict__ s, _Float16* __restrict__ d, int n) {
  for (int i = blockIdx.x * blockDim.x + threadIdx.x; i < n; i += gridDim.x * blockDim.x)
    d[i] = (_Float16)s[i];
}

// ---------------------------------------------------------------------------
// Generic WMMA GEMM: C(M,N) = A(M,K) @ W(N,K)^T + bias, optional relu.
// ---------------------------------------------------------------------------
__global__ void wmma_gemm(const _Float16* __restrict__ A, int lda,
                          const _Float16* __restrict__ W, int ldw,
                          const float* __restrict__ bias,
                          float* __restrict__ C, int ldc,
                          int M, int N, int K, int relu)
{
  int lane = threadIdx.x & 31;
  int widx = blockIdx.x * (blockDim.x >> 5) + (threadIdx.x >> 5);
  int mtiles = M >> 4;
  if (widx >= mtiles * (N >> 6)) return;
  int mt = widx % mtiles, nt = widx / mtiles;
  int m0 = mt << 4, n0 = nt << 6;

  v8f acc[4] = {v8f{}, v8f{}, v8f{}, v8f{}};
  gemm_tile_acc(A, lda, W, ldw, m0, n0, K, lane, acc);

  int hh = lane >> 4, nl = lane & 15;
#pragma unroll
  for (int s = 0; s < 4; ++s) {
    int col = n0 + (s << 4) + nl;
    float bs = bias ? bias[col] : 0.f;
#pragma unroll
    for (int r = 0; r < 8; ++r) {
      int row = m0 + r + (hh << 3);
      float v = acc[s][r] + bs;
      if (relu) v = fmaxf(v, 0.f);
      C[(size_t)row * ldc + col] = v;
    }
  }
}

// ---------------------------------------------------------------------------
// Persistent LSTM layer. Grid MUST be <<<8, 256>>> (64 wave32s).
//
// Ownership remap: block bk owns hidden slice j in [bk*64, bk*64+64) across
// ALL FOUR gates (wave w: gate g = w>>1, cols g*512 + bk*64 .. +64, M-tile
// mt = w&1).  Consequences:
//   - gate tile (32 x 256) lives in LDS, never touches global memory
//   - cell state c lives in LDS for the whole layer
//   - i/f/g/o -> c,h update is block-local (__syncthreads only)
//   - ONE grid barrier per timestep (publish h16), not two
// ---------------------------------------------------------------------------
__global__ void lstm_layer(const _Float16* __restrict__ X, int T, int Dx,
                           const _Float16* __restrict__ Wih,
                           const _Float16* __restrict__ Whh,
                           const float* __restrict__ bih,
                           const float* __restrict__ bhh,
                           float* __restrict__ h32,
                           _Float16* __restrict__ h16,
                           _Float16* __restrict__ seq,
                           unsigned* bar)
{
  __shared__ float gates_s[BB][GS];    // 32 x 256 gate tile (padded stride)
  __shared__ float c_s[BB][64];        // owned cell-state slice
  __shared__ float bias_s[256];        // bih+bhh for owned gate columns

  int phase = 0;
  const int bk   = blockIdx.x;                  // 0..7 : hidden slice owner
  const int lane = threadIdx.x & 31;
  const int wl   = threadIdx.x >> 5;            // 0..7 wave in block
  const int m0   = (wl & 1) << 4;               // 2 M-tiles (B=32)
  const int g    = wl >> 1;                     // gate 0..3 (i,f,g,o)
  const int n0   = g * HD + bk * 64;            // global weight-row base
  const int Xlda = T * Dx;                      // batch row stride of X

  // ---- init: zero c (LDS), owned h16 slice (global), preload bias ----
  {
    int gcol = (threadIdx.x >> 6) * HD + bk * 64 + (threadIdx.x & 63);
    bias_s[threadIdx.x] = bih[gcol] + bhh[gcol];
    for (int e = threadIdx.x; e < BB * 64; e += 256) {
      int b = e >> 6, jj = e & 63;
      c_s[b][jj] = 0.f;
      h16[b * HD + bk * 64 + jj] = (_Float16)0.f;
    }
  }
  gsync(bar, 8, &phase);

  for (int tstep = 0; tstep < T; ++tstep) {
    // ---- phase A: gate GEMM (fused input + recurrent projection) ----
    v8f acc[4] = {v8f{}, v8f{}, v8f{}, v8f{}};
    const _Float16* Xt = X + tstep * Dx;
    gemm_tile_acc(Xt,  Xlda, Wih, Dx, m0, n0, Dx, lane, acc);
    gemm_tile_acc(h16, HD,   Whh, HD, m0, n0, HD, lane, acc);
    {
      int hh = lane >> 4, nl = lane & 15;
#pragma unroll
      for (int s = 0; s < 4; ++s) {
        int jj = (s << 4) + nl;                 // 0..63 within slice
#pragma unroll
        for (int r = 0; r < 8; ++r)
          gates_s[m0 + r + (hh << 3)][g * 64 + jj] = acc[s][r];
      }
    }
    __syncthreads();

    // ---- phase B: block-local LSTM cell update on owned slice ----
    {
      int jj = threadIdx.x & 63;
      int j  = bk * 64 + jj;
      for (int b = threadIdx.x >> 6; b < BB; b += 4) {
        float gi = gates_s[b][jj]        + bias_s[jj];
        float gf = gates_s[b][64 + jj]   + bias_s[64 + jj];
        float gg = gates_s[b][128 + jj]  + bias_s[128 + jj];
        float go = gates_s[b][192 + jj]  + bias_s[192 + jj];
        float cp = c_s[b][jj];
        float cn = sigmf(gf) * cp + sigmf(gi) * tanhf(gg);
        float hn = sigmf(go) * tanhf(cn);
        c_s[b][jj] = cn;
        h32[b * HD + j] = hn;
        h16[b * HD + j] = (_Float16)hn;
        seq[((size_t)b * T + tstep) * HD + j] = (_Float16)hn;
      }
    }
    gsync(bar, 8, &phase);   // publish h16 grid-wide for next step
  }
}

// ---------------------------------------------------------------------------
// Attention scores + softmax for decoder step s = DT-1 only (the only step
// the output depends on). One block per batch b; thread e in [0,256).
// score[b,e] = sum_k tanh(pd[b,k] + pe[b*256+e,k]) * W2[k] + b2
// ---------------------------------------------------------------------------
__global__ void attn_scores(const float* __restrict__ pd,
                            const float* __restrict__ pe,
                            const float* __restrict__ W2,
                            const float* __restrict__ b2,
                            float* __restrict__ wout)
{
  __shared__ float pdS[HD];
  __shared__ float w2S[HD];
  __shared__ float red[256];
  int b = blockIdx.x, e = threadIdx.x;
  for (int k = e; k < HD; k += 256) { pdS[k] = pd[b * HD + k]; w2S[k] = W2[k]; }
  __syncthreads();

  const float* per = pe + ((size_t)(b * 256 + e)) * HD;
  float s = b2[0];
  for (int k = 0; k < HD; ++k) s += tanhf(pdS[k] + per[k]) * w2S[k];

  red[e] = s; __syncthreads();
  for (int o = 128; o > 0; o >>= 1) { if (e < o) red[e] = fmaxf(red[e], red[e + o]); __syncthreads(); }
  float mx = red[0]; __syncthreads();
  float ex = expf(s - mx);
  red[e] = ex; __syncthreads();
  for (int o = 128; o > 0; o >>= 1) { if (e < o) red[e] += red[e + o]; __syncthreads(); }
  wout[b * 256 + e] = ex / red[0];
}

// ---------------------------------------------------------------------------
// ctx[b,h] = sum_e w[b,e] * enc[(b*16+15)%32, e, h]  (torch repeat quirk),
// then comb = [dec_last_f32, ctx] as f16 (B, 1024).
// ---------------------------------------------------------------------------
__global__ void ctx_comb(const float* __restrict__ watt,
                         const _Float16* __restrict__ enc,
                         const float* __restrict__ hdec,
                         _Float16* __restrict__ comb)
{
  __shared__ float wS[256];
  int b = blockIdx.x, t = threadIdx.x;
  wS[t] = watt[b * 256 + t];
  __syncthreads();
  int idx = (b * DTT + (DTT - 1)) & (BB - 1);
  const _Float16* eb = enc + (size_t)idx * ETT * HD;
  for (int h = t; h < HD; h += 256) {
    float acc = 0.f;
    for (int e = 0; e < ETT; ++e) acc += wS[e] * (float)eb[(size_t)e * HD + h];
    comb[(size_t)b * 2 * HD + HD + h] = (_Float16)acc;
    comb[(size_t)b * 2 * HD + h]      = (_Float16)hdec[b * HD + h];
  }
}

// ---------------------------------------------------------------------------
// Final tiny GEMM: out(B,8) = proc(B,512) @ out_W(8,512)^T + out_b (fp32)
// ---------------------------------------------------------------------------
__global__ void out_final(const float* __restrict__ proc,
                          const float* __restrict__ outW,
                          const float* __restrict__ outb,
                          float* __restrict__ out)
{
  int i = threadIdx.x;                 // 256 = 32*8
  int b = i >> 3, o = i & 7;
  float s = outb[o];
  const float* p = proc + (size_t)b * HD;
  const float* wr = outW + (size_t)o * HD;
  for (int k = 0; k < HD; ++k) s += p[k] * wr[k];
  out[i] = s;
}

// ---------------------------------------------------------------------------
extern "C" void kernel_launch(void* const* d_in, const int* in_sizes, int n_in,
                              void* d_out, int out_size, void* d_ws, size_t ws_size,
                              hipStream_t stream)
{
  (void)in_sizes; (void)n_in; (void)out_size; (void)ws_size;

  const float* hf      = (const float*)d_in[0];
  const float* lf      = (const float*)d_in[1];
  const float* eWih0   = (const float*)d_in[2];
  const float* eWhh0   = (const float*)d_in[3];
  const float* ebih0   = (const float*)d_in[4];
  const float* ebhh0   = (const float*)d_in[5];
  const float* eWih1   = (const float*)d_in[6];
  const float* eWhh1   = (const float*)d_in[7];
  const float* ebih1   = (const float*)d_in[8];
  const float* ebhh1   = (const float*)d_in[9];
  const float* dWih0   = (const float*)d_in[10];
  const float* dWhh0   = (const float*)d_in[11];
  const float* dbih0   = (const float*)d_in[12];
  const float* dbhh0   = (const float*)d_in[13];
  const float* dWih1   = (const float*)d_in[14];
  const float* dWhh1   = (const float*)d_in[15];
  const float* dbih1   = (const float*)d_in[16];
  const float* dbhh1   = (const float*)d_in[17];
  const float* attW1   = (const float*)d_in[18];
  const float* attb1   = (const float*)d_in[19];
  const float* attW2   = (const float*)d_in[20];
  const float* attb2   = (const float*)d_in[21];
  const float* intW    = (const float*)d_in[22];
  const float* intb    = (const float*)d_in[23];
  const float* outW    = (const float*)d_in[24];
  const float* outb    = (const float*)d_in[25];

  // ---- workspace layout ----
  char* base = (char*)d_ws;
  size_t off = 0;
  auto alc = [&](size_t bytes) -> void* {
    void* p = base + off;
    off = (off + bytes + 255) & ~(size_t)255;
    return p;
  };
  _Float16* hf16    = (_Float16*)alc((size_t)BB * ETT * HFIN * 2);
  _Float16* lf16    = (_Float16*)alc((size_t)BB * DTT * LFIN * 2);
  _Float16* eWih0h  = (_Float16*)alc((size_t)G4 * HFIN * 2);
  _Float16* eWhh0h  = (_Float16*)alc((size_t)G4 * HD * 2);
  _Float16* eWih1h  = (_Float16*)alc((size_t)G4 * HD * 2);
  _Float16* eWhh1h  = (_Float16*)alc((size_t)G4 * HD * 2);
  _Float16* dWih0h  = (_Float16*)alc((size_t)G4 * LFIN * 2);
  _Float16* dWhh0h  = (_Float16*)alc((size_t)G4 * HD * 2);
  _Float16* dWih1h  = (_Float16*)alc((size_t)G4 * HD * 2);
  _Float16* dWhh1h  = (_Float16*)alc((size_t)G4 * HD * 2);
  _Float16* attW1h  = (_Float16*)alc((size_t)HD * 2 * HD * 2);
  _Float16* intWh   = (_Float16*)alc((size_t)HD * 2 * HD * 2);
  _Float16* h0seq   = (_Float16*)alc((size_t)BB * ETT * HD * 2);
  _Float16* encseq  = (_Float16*)alc((size_t)BB * ETT * HD * 2);
  _Float16* d0seq   = (_Float16*)alc((size_t)BB * DTT * HD * 2);
  _Float16* decseq  = (_Float16*)alc((size_t)BB * DTT * HD * 2);
  _Float16* h16     = (_Float16*)alc((size_t)BB * HD * 2);
  _Float16* comb16  = (_Float16*)alc((size_t)BB * 2 * HD * 2);
  float*    h32     = (float*)alc((size_t)BB * HD * 4);
  float*    pe      = (float*)alc((size_t)BB * ETT * HD * 4);
  float*    pd      = (float*)alc((size_t)BB * HD * 4);
  float*    watt    = (float*)alc((size_t)BB * ETT * 4);
  float*    proc    = (float*)alc((size_t)BB * HD * 4);
  unsigned* bars    = (unsigned*)alc(4 * sizeof(unsigned));

  // ---- f32 -> f16 conversions ----
  auto cvt = [&](const float* s, _Float16* d, size_t n) {
    int blocks = (int)((n + 255) / 256); if (blocks > 2048) blocks = 2048;
    cvt_f32_f16<<<blocks, 256, 0, stream>>>(s, d, (int)n);
  };
  cvt(hf,    hf16,   (size_t)BB * ETT * HFIN);
  cvt(lf,    lf16,   (size_t)BB * DTT * LFIN);
  cvt(eWih0, eWih0h, (size_t)G4 * HFIN);
  cvt(eWhh0, eWhh0h, (size_t)G4 * HD);
  cvt(eWih1, eWih1h, (size_t)G4 * HD);
  cvt(eWhh1, eWhh1h, (size_t)G4 * HD);
  cvt(dWih0, dWih0h, (size_t)G4 * LFIN);
  cvt(dWhh0, dWhh0h, (size_t)G4 * HD);
  cvt(dWih1, dWih1h, (size_t)G4 * HD);
  cvt(dWhh1, dWhh1h, (size_t)G4 * HD);
  cvt(attW1, attW1h, (size_t)HD * 2 * HD);
  cvt(intW,  intWh,  (size_t)HD * 2 * HD);

  hipMemsetAsync(bars, 0, 4 * sizeof(unsigned), stream);

  // ---- LSTM layers (persistent, 8 blocks x 256 threads) ----
  lstm_layer<<<8, 256, 0, stream>>>(hf16,  ETT, HFIN, eWih0h, eWhh0h, ebih0, ebhh0,
                                    h32, h16, h0seq, bars + 0);
  lstm_layer<<<8, 256, 0, stream>>>(h0seq, ETT, HD,   eWih1h, eWhh1h, ebih1, ebhh1,
                                    h32, h16, encseq, bars + 1);
  lstm_layer<<<8, 256, 0, stream>>>(lf16,  DTT, LFIN, dWih0h, dWhh0h, dbih0, dbhh0,
                                    h32, h16, d0seq, bars + 2);
  lstm_layer<<<8, 256, 0, stream>>>(d0seq, DTT, HD,   dWih1h, dWhh1h, dbih1, dbhh1,
                                    h32, h16, decseq, bars + 3);
  // after this, h32 = decoder final hidden state (f32)

  auto gemm = [&](const _Float16* A, int lda, const _Float16* W, int ldw,
                  const float* bias, float* C, int ldc, int M, int N, int K, int relu) {
    int tw = (M / 16) * (N / 64);
    wmma_gemm<<<(tw + 7) / 8, 256, 0, stream>>>(A, lda, W, ldw, bias, C, ldc, M, N, K, relu);
  };

  // pe = enc @ att_W1[:, H:]^T + att_b1   (B*E, H)
  gemm(encseq, HD, attW1h + HD, 2 * HD, attb1, pe, HD, BB * ETT, HD, HD, 0);
  // pd = dec[:, -1, :] @ att_W1[:, :H]^T  (B, H)
  gemm(decseq + (DTT - 1) * HD, DTT * HD, attW1h, 2 * HD, nullptr, pd, HD, BB, HD, HD, 0);

  attn_scores<<<BB, 256, 0, stream>>>(pd, pe, attW2, attb2, watt);
  ctx_comb<<<BB, 256, 0, stream>>>(watt, encseq, h32, comb16);

  // proc = relu(comb @ int_W^T + int_b)   (B, H)
  gemm(comb16, 2 * HD, intWh, 2 * HD, intb, proc, HD, BB, HD, 2 * HD, 1);

  out_final<<<1, 256, 0, stream>>>(proc, outW, outb, (float*)d_out);
}
